// VTF_SA_81063212745095
// MI455X (gfx1250) — matrix operand order repaired and checked by reading, not compiled, not run on hardware
//
#include <hip/hip_runtime.h>
#include <hip/hip_bf16.h>
#include <stdint.h>

typedef _Float16 h16;
typedef h16   v16h __attribute__((ext_vector_type(16)));
typedef h16   v8h  __attribute__((ext_vector_type(8)));
typedef float v8f  __attribute__((ext_vector_type(8)));

#define C_DIM 1024
#define CKD   128
#define NTOK  1024
#define BATCH 16

// ---------------- pack Wq/Wk (fp32 -> f16, stacked [256][1024]) ----------------
__global__ void k_pack_w(const float* __restrict__ Wq, const float* __restrict__ Wk,
                         h16* __restrict__ Wqk) {
    for (int i = blockIdx.x * blockDim.x + threadIdx.x; i < 256 * C_DIM;
         i += gridDim.x * blockDim.x) {
        int r = i >> 10, c = i & (C_DIM - 1);
        float v = (r < CKD) ? Wq[r * C_DIM + c] : Wk[(r - CKD) * C_DIM + c];
        Wqk[i] = (h16)v;
    }
}

// ---------------- transpose x [B,C,N] fp32 -> xbT [B,N,C] f16 ----------------
__global__ void k_transpose(const float* __restrict__ x, h16* __restrict__ xbT) {
    __shared__ h16 tile[32][34];
    int b = blockIdx.z;
    int c0 = blockIdx.y * 32, n0 = blockIdx.x * 32;
    int tx = threadIdx.x, ty = threadIdx.y;      // block (32,8)
    const float* xb = x + (size_t)b * C_DIM * NTOK;
    #pragma unroll
    for (int i = 0; i < 4; i++) {
        int c = c0 + ty + i * 8;
        tile[ty + i * 8][tx] = (h16)xb[(size_t)c * NTOK + n0 + tx];
    }
    __syncthreads();
    h16* outp = xbT + (size_t)b * NTOK * C_DIM;
    #pragma unroll
    for (int i = 0; i < 4; i++) {
        int n = n0 + ty + i * 8;
        outp[(size_t)n * C_DIM + c0 + tx] = tile[tx][ty + i * 8];
    }
}

// ---------------- projection: [256,1024] x [1024,1024] per batch (WMMA) ----------------
// A = Wqk row-major, B = xbT (pre-transposed: lane fragment = 2 x b128).
// D stored transposed with bias: qT/kT [B][N][128] f16 (A/B-ready for energy GEMM).
__global__ void __launch_bounds__(256) k_proj(const h16* __restrict__ Wqk,
                                              const h16* __restrict__ xbT,
                                              const float* __restrict__ bq,
                                              const float* __restrict__ bk,
                                              h16* __restrict__ qT, h16* __restrict__ kT) {
    int b    = blockIdx.z;
    int lane = threadIdx.x & 31, wave = threadIdx.x >> 5;
    int wm = wave >> 2, wn = wave & 3;               // 2 (M) x 4 (N) waves
    int m0 = blockIdx.y * 64 + wm * 32;              // wave: 32 M rows (2 tiles)
    int n0 = blockIdx.x * 256 + wn * 64;             // wave: 64 N cols (4 tiles)
    int half = lane >> 4, lr = lane & 15;
    const h16* Bb = xbT + (size_t)b * NTOK * C_DIM;

    v8f acc[2][4] = {};
    for (int kb = 0; kb < C_DIM; kb += 32) {
        v16h afrag[2];
        #pragma unroll
        for (int r = 0; r < 2; r++) {
            const h16* pa = Wqk + (size_t)(m0 + r * 16 + lr) * C_DIM + kb + half * 16;
            ((v8h*)&afrag[r])[0] = *(const v8h*)pa;
            ((v8h*)&afrag[r])[1] = *(const v8h*)(pa + 8);
        }
        #pragma unroll
        for (int t = 0; t < 4; t++) {
            const h16* pb = Bb + (size_t)(n0 + t * 16 + lr) * C_DIM + kb + half * 16;
            v16h bfrag;
            ((v8h*)&bfrag)[0] = *(const v8h*)pb;
            ((v8h*)&bfrag)[1] = *(const v8h*)(pb + 8);
            #pragma unroll
            for (int r = 0; r < 2; r++)
                acc[r][t] = __builtin_amdgcn_wmma_f32_16x16x32_f16(
                    false, afrag[r], false, bfrag, (short)0, acc[r][t], false, false);
        }
    }
    // store transposed with bias: lane holds col n = tile_n + lr, rows m = mbase..mbase+7 contiguous
    #pragma unroll
    for (int r = 0; r < 2; r++) {
        int mbase = m0 + r * 16 + half * 8;
        bool isq = (mbase < CKD);
        const float* bias = isq ? (bq + mbase) : (bk + (mbase - CKD));
        h16* dst = isq ? (qT + (size_t)b * NTOK * CKD + mbase)
                       : (kT + (size_t)b * NTOK * CKD + (mbase - CKD));
        #pragma unroll
        for (int t = 0; t < 4; t++) {
            int n = n0 + t * 16 + lr;
            v8h o;
            #pragma unroll
            for (int v = 0; v < 8; v++) o[v] = (h16)(acc[r][t][v] + bias[v]);
            *(v8h*)(dst + (size_t)n * CKD) = o;
        }
    }
}

// ---------------- energy + softmax + column means (WMMA) ----------------
// strip of 16 rows (n) x 1024 cols (m) per workgroup; wave w owns cols w*128..+127.
__global__ void __launch_bounds__(256) k_energy(const h16* __restrict__ qT,
                                                const h16* __restrict__ kT,
                                                float* __restrict__ abar) {
    int b  = blockIdx.y;
    int n0 = blockIdx.x * 16;
    int lane = threadIdx.x & 31, wave = threadIdx.x >> 5;
    int half = lane >> 4, lr = lane & 15;
    const h16* Ab = qT + (size_t)b * NTOK * CKD;
    const h16* Bb = kT + (size_t)b * NTOK * CKD;

    v8f acc[8] = {};
    #pragma unroll
    for (int kb = 0; kb < CKD; kb += 32) {
        v16h afrag;
        const h16* pa = Ab + (size_t)(n0 + lr) * CKD + kb + half * 16;
        ((v8h*)&afrag)[0] = *(const v8h*)pa;
        ((v8h*)&afrag)[1] = *(const v8h*)(pa + 8);
        #pragma unroll
        for (int t = 0; t < 8; t++) {
            int m = wave * 128 + t * 16 + lr;
            const h16* pb = Bb + (size_t)m * CKD + kb + half * 16;
            v16h bfrag;
            ((v8h*)&bfrag)[0] = *(const v8h*)pb;
            ((v8h*)&bfrag)[1] = *(const v8h*)(pb + 8);
            acc[t] = __builtin_amdgcn_wmma_f32_16x16x32_f16(
                false, afrag, false, bfrag, (short)0, acc[t], false, false);
        }
    }

    __shared__ float red[8][16];
    __shared__ float rstat[16];

    // row max: lane's VGPR v <-> row half*8+v; reduce over 8 tiles then 16-lane half
    float rm[8];
    #pragma unroll
    for (int v = 0; v < 8; v++) {
        float m = acc[0][v];
        #pragma unroll
        for (int t = 1; t < 8; t++) m = fmaxf(m, acc[t][v]);
        #pragma unroll
        for (int off = 1; off < 16; off <<= 1) m = fmaxf(m, __shfl_xor(m, off, 32));
        rm[v] = m;
    }
    if (lr == 0) {
        #pragma unroll
        for (int v = 0; v < 8; v++) red[wave][half * 8 + v] = rm[v];
    }
    __syncthreads();
    if (threadIdx.x < 16) {
        float m = red[0][threadIdx.x];
        for (int w = 1; w < 8; w++) m = fmaxf(m, red[w][threadIdx.x]);
        rstat[threadIdx.x] = m;
    }
    __syncthreads();
    #pragma unroll
    for (int v = 0; v < 8; v++) rm[v] = rstat[half * 8 + v];
    __syncthreads();

    // exp + row sum
    float rs[8];
    #pragma unroll
    for (int v = 0; v < 8; v++) {
        float s = 0.f;
        #pragma unroll
        for (int t = 0; t < 8; t++) {
            float e = __builtin_amdgcn_exp2f((acc[t][v] - rm[v]) * 1.4426950408889634f);
            acc[t][v] = e;
            s += e;
        }
        #pragma unroll
        for (int off = 1; off < 16; off <<= 1) s += __shfl_xor(s, off, 32);
        rs[v] = s;
    }
    if (lr == 0) {
        #pragma unroll
        for (int v = 0; v < 8; v++) red[wave][half * 8 + v] = rs[v];
    }
    __syncthreads();
    if (threadIdx.x < 16) {
        float s = 0.f;
        for (int w = 0; w < 8; w++) s += red[w][threadIdx.x];
        rstat[threadIdx.x] = __builtin_amdgcn_rcpf(s);
    }
    __syncthreads();
    float inv[8];
    #pragma unroll
    for (int v = 0; v < 8; v++) inv[v] = rstat[half * 8 + v];

    // column partial sums (over the 16 rows of this strip) -> abar (scaled by 1/N)
    #pragma unroll
    for (int t = 0; t < 8; t++) {
        float cs = 0.f;
        #pragma unroll
        for (int v = 0; v < 8; v++) cs += acc[t][v] * inv[v];
        cs += __shfl_xor(cs, 16, 32);   // combine row-halves (same column m)
        if (half == 0)
            atomicAdd(&abar[(size_t)b * NTOK + wave * 128 + t * 16 + lr],
                      cs * (1.0f / 1024.0f));
    }
}

// ---------------- t[b,c] = sum_n x[b,c,n]*abar[b,n] ; xbar[b,c] = mean_n x ----------------
__global__ void __launch_bounds__(256) k_wsum(const float* __restrict__ x,
                                              const float* __restrict__ abar,
                                              float* __restrict__ tvec,
                                              float* __restrict__ xbar) {
    int b = blockIdx.y;
    int c = blockIdx.x * 8 + (threadIdx.x >> 5);
    int lane = threadIdx.x & 31;
    const float* xr = x + ((size_t)b * C_DIM + c) * NTOK;
    const float* ar = abar + (size_t)b * NTOK;
    float s = 0.f, xs = 0.f;
    for (int n = lane; n < NTOK; n += 32) {
        float xv = xr[n];
        s  += xv * ar[n];
        xs += xv;
    }
    #pragma unroll
    for (int off = 1; off < 32; off <<= 1) {
        s  += __shfl_xor(s, off, 32);
        xs += __shfl_xor(xs, off, 32);
    }
    if (lane == 0) {
        tvec[(size_t)b * C_DIM + c] = s;
        xbar[(size_t)b * C_DIM + c] = xs * (1.0f / 1024.0f);
    }
}

// ---------------- pooled = gamma*(Wv.t + bv) + xbar ----------------
__global__ void __launch_bounds__(256) k_pool(const float* __restrict__ Wv,
                                              const float* __restrict__ bv,
                                              const float* __restrict__ tvec,
                                              const float* __restrict__ xbar,
                                              const float* __restrict__ gamma,
                                              float* __restrict__ pooled) {
    int b = blockIdx.y;
    int c = blockIdx.x * 8 + (threadIdx.x >> 5);
    int lane = threadIdx.x & 31;
    const float* wr = Wv + (size_t)c * C_DIM;
    const float* tr = tvec + (size_t)b * C_DIM;
    float s = 0.f;
    for (int i = lane; i < C_DIM; i += 32) s += wr[i] * tr[i];
    #pragma unroll
    for (int off = 1; off < 32; off <<= 1) s += __shfl_xor(s, off, 32);
    if (lane == 0)
        pooled[(size_t)b * C_DIM + c] =
            gamma[0] * (s + bv[c]) + xbar[(size_t)b * C_DIM + c];
}

// ---------------- fc1 ----------------
__global__ void __launch_bounds__(256) k_fc1(const float* __restrict__ pooled,
                                             const float* __restrict__ W1,
                                             const float* __restrict__ b1,
                                             float* __restrict__ hdn) {
    int idx = blockIdx.x * 8 + (threadIdx.x >> 5);  // b*64 + h
    int b = idx >> 6, h = idx & 63;
    int lane = threadIdx.x & 31;
    const float* pr = pooled + (size_t)b * C_DIM;
    const float* wr = W1 + (size_t)h * C_DIM;
    float s = 0.f;
    for (int i = lane; i < C_DIM; i += 32) s += pr[i] * wr[i];
    #pragma unroll
    for (int off = 1; off < 32; off <<= 1) s += __shfl_xor(s, off, 32);
    if (lane == 0) hdn[idx] = s + b1[h];
}

// ---------------- fc2 ----------------
__global__ void k_fc2(const float* __restrict__ hdn, const float* __restrict__ W2,
                      const float* __restrict__ b2, float* __restrict__ outp) {
    int i = threadIdx.x;             // 32 outputs: b = i>>1, o = i&1
    int b = i >> 1, o = i & 1;
    float s = 0.f;
    for (int h = 0; h < 64; h++) s += hdn[b * 64 + h] * W2[o * 64 + h];
    outp[i] = s + b2[o];
}

extern "C" void kernel_launch(void* const* d_in, const int* in_sizes, int n_in,
                              void* d_out, int out_size, void* d_ws, size_t ws_size,
                              hipStream_t stream) {
    const float* x  = (const float*)d_in[0];
    const float* Wq = (const float*)d_in[1];
    const float* bq = (const float*)d_in[2];
    const float* Wk = (const float*)d_in[3];
    const float* bk = (const float*)d_in[4];
    const float* Wv = (const float*)d_in[5];
    const float* bv = (const float*)d_in[6];
    const float* gamma = (const float*)d_in[7];
    const float* W1 = (const float*)d_in[8];
    const float* b1 = (const float*)d_in[9];
    const float* W2 = (const float*)d_in[10];
    const float* b2 = (const float*)d_in[11];
    float* outp = (float*)d_out;

    char* ws = (char*)d_ws;
    size_t off = 0;
    auto take = [&](size_t bytes) -> char* {
        char* p = ws + off;
        off = (off + bytes + 255) & ~(size_t)255;
        return p;
    };
    h16*   xbT    = (h16*)take((size_t)BATCH * NTOK * C_DIM * 2);  // 32 MB
    h16*   Wqk    = (h16*)take((size_t)256 * C_DIM * 2);           // 512 KB
    h16*   qT     = (h16*)take((size_t)BATCH * NTOK * CKD * 2);    // 4 MB
    h16*   kT     = (h16*)take((size_t)BATCH * NTOK * CKD * 2);    // 4 MB
    float* abar   = (float*)take((size_t)BATCH * NTOK * 4);
    float* tvec   = (float*)take((size_t)BATCH * C_DIM * 4);
    float* xbar   = (float*)take((size_t)BATCH * C_DIM * 4);
    float* pooled = (float*)take((size_t)BATCH * C_DIM * 4);
    float* hdn    = (float*)take((size_t)BATCH * 64 * 4);

    hipMemsetAsync(abar, 0, (size_t)BATCH * NTOK * 4, stream);

    k_pack_w<<<256, 256, 0, stream>>>(Wq, Wk, Wqk);
    k_transpose<<<dim3(32, 32, BATCH), dim3(32, 8), 0, stream>>>(x, xbT);
    k_proj<<<dim3(4, 4, BATCH), 256, 0, stream>>>(Wqk, xbT, bq, bk, qT, kT);
    k_energy<<<dim3(64, BATCH), 256, 0, stream>>>(qT, kT, abar);
    k_wsum<<<dim3(128, BATCH), 256, 0, stream>>>(x, abar, tvec, xbar);
    k_pool<<<dim3(128, BATCH), 256, 0, stream>>>(Wv, bv, tvec, xbar, gamma, pooled);
    k_fc1<<<128, 256, 0, stream>>>(pooled, W1, b1, hdn);
    k_fc2<<<1, 32, 0, stream>>>(hdn, W2, b2, outp);
}
